// GemResNetBlock_82222853915380
// MI455X (gfx1250) — matrix operand order repaired and checked by reading, not compiled, not run on hardware
//
#include <hip/hip_runtime.h>
#include <stdint.h>

#define NUM_V   20000
#define N_EDGES 160000
#define IN_CH   16
#define OUT_CH  32
#define NB      19
#define DIM     5      // 2*order+1
#define NFR     10     // 5 freqs * 2 rings
#define NCOL    160    // OUT_CH * DIM
#define NSAMP   7
#define PI_F    3.14159265358979323846f

typedef __attribute__((ext_vector_type(16))) _Float16 v16h;
typedef __attribute__((ext_vector_type(8)))  float    v8f;
typedef __attribute__((ext_vector_type(4)))  uint32_t u32x4;

// GCC-style vector + explicit address spaces to match the async builtin's
// prototype exactly: void(v4i as1*, v4i as3*, Ii, Ii).
typedef int v4i __attribute__((vector_size(4 * sizeof(int))));
typedef __attribute__((address_space(1))) v4i glb_v4i;
typedef __attribute__((address_space(3))) v4i lds_v4i;

// Async global->LDS (gfx1250) if the toolchain exposes it; safe fallback otherwise.
#if defined(__has_builtin)
#  if __has_builtin(__builtin_amdgcn_global_load_async_to_lds_b128) && \
      __has_builtin(__builtin_amdgcn_s_wait_asynccnt)
#    define HAVE_ASYNC_LDS 1
#  endif
#endif
#ifndef HAVE_ASYNC_LDS
#  define HAVE_ASYNC_LDS 0
#endif

// ---------------------------------------------------------------------------
// Equivariant kernel basis (build_kernel_basis), encoded as 19 sparse blocks.
// ---------------------------------------------------------------------------
struct BasisDesc { int8_t fb, r0, c0, nr, nc; int8_t C[4]; int8_t S[4]; };
__constant__ BasisDesc g_basis[NB] = {
  {0,0,0,1,1,{ 1, 0, 0, 0},{ 0, 0, 0, 0}},  // (0,0)
  {1,0,1,1,2,{ 1, 0, 0, 0},{ 0, 1, 0, 0}},  // (0,1) a
  {1,0,1,1,2,{ 0, 1, 0, 0},{-1, 0, 0, 0}},  // (0,1) b
  {2,0,3,1,2,{ 1, 0, 0, 0},{ 0, 1, 0, 0}},  // (0,2) a
  {2,0,3,1,2,{ 0, 1, 0, 0},{-1, 0, 0, 0}},  // (0,2) b
  {1,1,0,2,1,{ 1, 0, 0, 0},{ 0, 1, 0, 0}},  // (1,0) a
  {1,1,0,2,1,{ 0, 1, 0, 0},{-1, 0, 0, 0}},  // (1,0) b
  {0,1,1,2,2,{ 1, 0, 0, 1},{ 0, 0, 0, 0}},  // (1,1) f=0 I
  {0,1,1,2,2,{ 0,-1, 1, 0},{ 0, 0, 0, 0}},  // (1,1) f=0 E
  {2,1,1,2,2,{ 1, 0, 0,-1},{ 0, 1, 1, 0}},  // (1,1) f=2 S,ES
  {2,1,1,2,2,{ 0, 1, 1, 0},{-1, 0, 0, 1}},  // (1,1) f=2 ES,-S
  {1,1,3,2,2,{ 1, 0, 0, 1},{ 0, 1,-1, 0}},  // (1,2) I,-E
  {1,1,3,2,2,{ 0,-1, 1, 0},{ 1, 0, 0, 1}},  // (1,2) E, I
  {2,3,0,2,1,{ 1, 0, 0, 0},{ 0, 1, 0, 0}},  // (2,0) a
  {2,3,0,2,1,{ 0, 1, 0, 0},{-1, 0, 0, 0}},  // (2,0) b
  {1,3,1,2,2,{ 1, 0, 0, 1},{ 0,-1, 1, 0}},  // (2,1) I, E
  {1,3,1,2,2,{ 0,-1, 1, 0},{-1, 0, 0,-1}},  // (2,1) E,-I
  {0,3,3,2,2,{ 1, 0, 0, 1},{ 0, 0, 0, 0}},  // (2,2) f=0 I
  {0,3,3,2,2,{ 0,-1, 1, 0},{ 0, 0, 0, 0}},  // (2,2) f=0 E
};

__device__ __forceinline__ float basisK(int b, int fq, int i, int j) {
  const BasisDesc& B = g_basis[b];
  if (i < B.r0 || i >= B.r0 + B.nr || j < B.c0 || j >= B.c0 + B.nc) return 0.f;
  int idx = (i - B.r0) * B.nc + (j - B.c0);
  if (B.fb == 0) return (fq == 0) ? (float)B.C[idx] : 0.f;
  if (fq == 2 * B.fb - 1) return (float)B.C[idx];
  if (fq == 2 * B.fb)     return (float)B.S[idx];
  return 0.f;
}

// ---------------------------------------------------------------------------
// wk[(f,r,c,j),(o,i)] = sum_b K[b,f,i,j] * W[b,r,o,c], stored f16 and
// PRE-SWIZZLED into WMMA B-fragment tiles:
//   tileIdx = nt*KT + kt  (kt = k/32, KT = KD/32)
//   tile[lane][h] = B[kt*32 + (lane>>4)*16 + h][nt*16 + (lane&15)],  h=0..15
// ---------------------------------------------------------------------------
template<int CIN>
__global__ void build_wk_kernel(const float* __restrict__ W, _Float16* __restrict__ wk) {
  const int KD = NFR * CIN * DIM;
  const int KT = KD / 32;
  int idx = blockIdx.x * 256 + threadIdx.x;
  if (idx >= KD * NCOL) return;
  int n = idx % NCOL;
  int k = idx / NCOL;
  int o = n / DIM, i = n % DIM;
  int fr = k / (CIN * DIM);
  int rem = k % (CIN * DIM);
  int c = rem / DIM, j = rem % DIM;
  int f = fr >> 1, r = fr & 1;
  float acc = 0.f;
#pragma unroll
  for (int b = 0; b < NB; ++b) {
    float kb = basisK(b, f, i, j);
    acc += kb * W[((b * 2 + r) * OUT_CH + o) * CIN + c];
  }
  // swizzled store
  int kt = k >> 5, kr = k & 31;
  int hi = kr >> 4, h = kr & 15;
  int nt = n >> 4, ncol = n & 15;
  int lane = hi * 16 + ncol;
  wk[((nt * KT + kt) * 32 + lane) * 16 + h] = (_Float16)acc;
}

// ---------------------------------------------------------------------------
// Fused gem_conv: gather + gauge transport + (z = precomp (x) xt) GEMM via
// v_wmma_f32_16x16x32_f16 + atomic scatter-add over dst.
// Block = 256 threads (8 wave32), 128 edges. K-chunk outer (A frags built
// once per chunk), N-tile inner with 10 persistent accumulator sets.
// ---------------------------------------------------------------------------
template<int CIN>
__global__ __launch_bounds__(256)
void conv_kernel(const float* __restrict__ feat,       // [NUM_V, CIN, 5]
                 const int*   __restrict__ dst,
                 const int*   __restrict__ src,
                 const float* __restrict__ precomp,    // [E, 5, 2]
                 const float* __restrict__ conn,       // [E]
                 const _Float16* __restrict__ wk,      // swizzled, KD*160 halves
                 float* __restrict__ outv)             // [NUM_V, 160]
{
  constexpr int KD     = NFR * CIN * DIM;   // 800 / 1600
  constexpr int KT     = KD / 32;           // 25 / 50
  constexpr int KCHUNK = 160;
  constexpr int NCH    = KD / KCHUNK;       // 5 / 10
  constexpr int EPB    = 128;
  constexpr int NST    = KCHUNK / 32;       // 5 WMMA steps per chunk
  constexpr int CD     = CIN * DIM;         // 80 / 160
  constexpr int XSTR   = CD + 8;            // padded xt row (bank-conflict-free)

  __shared__ _Float16 sB [NST * 512];       // swizzled B tiles for one chunk (5 KB)
  __shared__ _Float16 sXT[EPB * XSTR];      // rotated features (f16), padded rows
  __shared__ float    sP [EPB * NFR];       // precomp per edge

  const int tid   = threadIdx.x;
  const int eBase = blockIdx.x * EPB;

  // Gather x[src], apply rep_act (rotate pairs by m*alpha), store f16.
  for (int idx = tid; idx < EPB * CIN; idx += 256) {
    int el = idx / CIN, c = idx % CIN;
    int e = eBase + el;
    int s = src[e];
    float a = conn[e];
    float c1, s1, c2, s2;
    __sincosf(a, &s1, &c1);
    __sincosf(2.f * a, &s2, &c2);
    const float* xp = feat + (s * CIN + c) * DIM;
    float x0 = xp[0], x1 = xp[1], x2 = xp[2], x3 = xp[3], x4 = xp[4];
    _Float16* op = sXT + el * XSTR + c * DIM;
    op[0] = (_Float16)x0;
    op[1] = (_Float16)(c1 * x1 - s1 * x2);
    op[2] = (_Float16)(s1 * x1 + c1 * x2);
    op[3] = (_Float16)(c2 * x3 - s2 * x4);
    op[4] = (_Float16)(s2 * x3 + c2 * x4);
  }
  for (int idx = tid; idx < EPB * NFR; idx += 256)
    sP[idx] = precomp[eBase * NFR + idx];

  const int lane = tid & 31;
  const int w    = tid >> 5;        // wave id: edges [w*16, w*16+16)
  const int hi   = lane >> 4;
  const int ncol = lane & 15;       // B/C column within N-tile
  const int eA   = w * 16 + (lane & 15);   // A row (edge within wave group)

  const _Float16* xrow = sXT + eA * XSTR;
  const float*    prow = sP  + eA * NFR;

  v8f acc[10] = {};

  for (int ch = 0; ch < NCH; ++ch) {
    // ---- build the 5 A fragments for this chunk (reused by 10 N-tiles)
    v16h afr[NST];
#pragma unroll
    for (int st = 0; st < NST; ++st) {
#pragma unroll
      for (int v = 0; v < 8; ++v) {
        // contiguous k order used here, same as swizzled B: lanes<16 -> K {0..15}, lanes>=16 -> {16..31}
        int koff = 16 * hi + 2 * v;
        int t = st * 32 + koff;            // 0..158, even
        int fr, rem;
        if (CD == KCHUNK) { fr = ch; rem = t; }
        else { int g = (t >= CD) ? 1 : 0; fr = ch * (KCHUNK / CD) + g; rem = t - g * CD; }
        float p = prow[fr];
        afr[st][2 * v]     = (_Float16)(p * (float)xrow[rem]);
        afr[st][2 * v + 1] = (_Float16)(p * (float)xrow[rem + 1]);
      }
    }

#pragma unroll
    for (int nt = 0; nt < NCOL / 16; ++nt) {
      __syncthreads();
      // ---- stage swizzled B tiles for (ch, nt): one contiguous 5 KB block
      {
        const _Float16* gsrc = wk + (size_t)(nt * KT + ch * NST) * 512;
#if HAVE_ASYNC_LDS
        for (int idx = tid; idx < NST * 512 / 8; idx += 256)
          __builtin_amdgcn_global_load_async_to_lds_b128(
              (glb_v4i*)(gsrc + idx * 8), (lds_v4i*)(sB + idx * 8), 0, 0);
        __builtin_amdgcn_s_wait_asynccnt(0);
#else
        const u32x4* g4 = (const u32x4*)gsrc;
        u32x4*       l4 = (u32x4*)sB;
        for (int idx = tid; idx < NST * 512 / 8; idx += 256)
          l4[idx] = g4[idx];
#endif
      }
      __syncthreads();
#pragma unroll
      for (int st = 0; st < NST; ++st) {
        // lane's fragment = 32 contiguous bytes -> 2x ds_load_b128
        const _Float16* bp = sB + st * 512 + lane * 16;
        union { u32x4 q[2]; v16h h; } bu;
        bu.q[0] = *(const u32x4*)bp;
        bu.q[1] = *(const u32x4*)(bp + 8);
        acc[nt] = __builtin_amdgcn_wmma_f32_16x16x32_f16(
            false, afr[st], false, bu.h, (short)0, acc[nt], false, false);
      }
    }
  }

  // ---- scatter-add: C/D layout: VGPR r holds M = r + 8*hi, column = lane&15
  int dstv[8];
#pragma unroll
  for (int r = 0; r < 8; ++r)
    dstv[r] = dst[eBase + w * 16 + r + hi * 8];
#pragma unroll
  for (int nt = 0; nt < NCOL / 16; ++nt) {
#pragma unroll
    for (int r = 0; r < 8; ++r)
      atomicAdd(outv + dstv[r] * NCOL + nt * 16 + ncol, acc[nt][r]);
  }
}

// ---------------------------------------------------------------------------
// regular_relu (order 2, 7 samples) helper + elementwise kernels
// ---------------------------------------------------------------------------
__device__ __forceinline__ void regular_relu5(const float* din, float* dout) {
  float o0 = 0, o1 = 0, o2 = 0, o3 = 0, o4 = 0;
#pragma unroll
  for (int s = 0; s < NSAMP; ++s) {
    float th = (2.0f * PI_F) * (float)s / (float)NSAMP;
    float c1, s1, c2, s2;
    __sincosf(th, &s1, &c1);
    __sincosf(2.f * th, &s2, &c2);
    float v = din[0] + c1 * din[1] + s1 * din[2] + c2 * din[3] + s2 * din[4];
    v = fmaxf(v, 0.f);
    o0 += v;
    o1 += 2.f * c1 * v; o2 += 2.f * s1 * v;
    o3 += 2.f * c2 * v; o4 += 2.f * s2 * v;
  }
  const float inv = 1.f / (float)NSAMP;
  dout[0] = o0 * inv; dout[1] = o1 * inv; dout[2] = o2 * inv;
  dout[3] = o3 * inv; dout[4] = o4 * inv;
}

__global__ void bias_relu_kernel(float* __restrict__ y, const float* __restrict__ b) {
  int idx = blockIdx.x * 256 + threadIdx.x;
  if (idx >= NUM_V * OUT_CH) return;
  int o = idx % OUT_CH;
  float* p = y + idx * DIM;
  float d[5], r[5];
#pragma unroll
  for (int t = 0; t < DIM; ++t) d[t] = p[t];
  d[0] += b[o];
  regular_relu5(d, r);
#pragma unroll
  for (int t = 0; t < DIM; ++t) p[t] = r[t];
}

__global__ void final_kernel(const float* __restrict__ y2, const float* __restrict__ b2,
                             const float* __restrict__ x,  const float* __restrict__ Wlin,
                             const float* __restrict__ blin, float* __restrict__ outp) {
  int idx = blockIdx.x * 256 + threadIdx.x;
  if (idx >= NUM_V * OUT_CH) return;
  int v = idx / OUT_CH, o = idx % OUT_CH;
  const float* p = y2 + idx * DIM;
  float d[5];
#pragma unroll
  for (int t = 0; t < DIM; ++t) d[t] = p[t];
  d[0] += b2[o];
  float bl = blin[o];
#pragma unroll
  for (int t = 0; t < DIM; ++t) {
    float r = bl;
#pragma unroll
    for (int c = 0; c < IN_CH; ++c)
      r += x[(v * IN_CH + c) * DIM + t] * Wlin[o * IN_CH + c];
    d[t] += r;
  }
  float out[5];
  regular_relu5(d, out);
  float* q = outp + idx * DIM;
#pragma unroll
  for (int t = 0; t < DIM; ++t) q[t] = out[t];
}

// ---------------------------------------------------------------------------
extern "C" void kernel_launch(void* const* d_in, const int* in_sizes, int n_in,
                              void* d_out, int out_size, void* d_ws, size_t ws_size,
                              hipStream_t stream) {
  const float* x       = (const float*)d_in[0];
  const int*   ei      = (const int*)  d_in[1];
  const float* precomp = (const float*)d_in[2];
  const float* conn    = (const float*)d_in[3];
  const float* W1      = (const float*)d_in[4];
  const float* b1      = (const float*)d_in[5];
  const float* W2      = (const float*)d_in[6];
  const float* b2      = (const float*)d_in[7];
  const float* Wlin    = (const float*)d_in[8];
  const float* blin    = (const float*)d_in[9];
  float* outp = (float*)d_out;

  char* ws = (char*)d_ws;
  _Float16* wk = (_Float16*)ws;                                    // <= 512 KB
  float* y1 = (float*)(ws + 512 * 1024);                           // 12.8 MB
  float* y2 = (float*)(ws + 512 * 1024 + 13 * 1024 * 1024);        // 12.8 MB

  const int* dst = ei;             // edge_index[0]
  const int* src = ei + N_EDGES;   // edge_index[1]

  (void)hipMemsetAsync(y1, 0, (size_t)NUM_V * NCOL * sizeof(float), stream);
  (void)hipMemsetAsync(y2, 0, (size_t)NUM_V * NCOL * sizeof(float), stream);

  // conv1: C_in = 16, K = 800
  build_wk_kernel<IN_CH><<<(NFR * IN_CH * DIM * NCOL + 255) / 256, 256, 0, stream>>>(W1, wk);
  conv_kernel<IN_CH><<<N_EDGES / 128, 256, 0, stream>>>(x, dst, src, precomp, conn, wk, y1);
  bias_relu_kernel<<<(NUM_V * OUT_CH + 255) / 256, 256, 0, stream>>>(y1, b1);

  // conv2: C_in = 32, K = 1600
  build_wk_kernel<OUT_CH><<<(NFR * OUT_CH * DIM * NCOL + 255) / 256, 256, 0, stream>>>(W2, wk);
  conv_kernel<OUT_CH><<<N_EDGES / 128, 256, 0, stream>>>(y1, dst, src, precomp, conn, wk, y2);

  // residual + bias + final regular_relu
  final_kernel<<<(NUM_V * OUT_CH + 255) / 256, 256, 0, stream>>>(y2, b2, x, Wlin, blin, outp);
}
